// PaulingNet_180388627168
// MI455X (gfx1250) — matrix operand order repaired and checked by reading, not compiled, not run on hardware
//
#include <hip/hip_runtime.h>
#include <hip/hip_bf16.h>
#include <math.h>

// ---------------------------------------------------------------------------
// PaulingNet forward (+ approximate analytic forces) for MI455X / gfx1250.
// wave32, WMMA bf16 16x16x32 with f32 accumulation. Memory-bound problem:
// all GEMM operands stored bf16, weights staged in LDS per block (shared by
// all 8 waves), epilogues fused, and the edge->atom reduction fused into one
// kernel to kill the 128 MB/iter dterm round-trip.
// ---------------------------------------------------------------------------

typedef __bf16 bf16;
typedef __attribute__((ext_vector_type(8)))  float v8f;
typedef __attribute__((ext_vector_type(8)))  bf16  v8bf;
typedef __attribute__((ext_vector_type(16))) bf16  v16bf;

#define NFc   128
#define RESc  20
#define Kc    64
#define Ac    256
#define Bc    8
#define RTOT  (Bc*Ac)        // 2048 atom rows
#define REDGE (Bc*Ac*Kc)     // 131072 edge rows

union Frag16 { v16bf v; v8bf h[2]; };

// ---------------------------------------------------------------------------
// WMMA GEMM: C[M,Nd] = act(A[M,Kd] @ W[Kd,Nd] + bias) (* rowscale)
// A row-major bf16 (lda); W pre-transposed WT[Nd][ldw] bf16, staged in LDS
// once per 256-thread block (8 waves share it). One wave computes a 16-row
// strip across all Nd (NT = Nd/16 accumulators), reusing its A fragment.
// LDS rows padded by +8 bf16 to spread bank accesses.
// ---------------------------------------------------------------------------
template<int NT>
__global__ void k_gemm(const bf16* __restrict__ A, int lda,
                       const bf16* __restrict__ WT, int ldw,
                       const float* __restrict__ bias,
                       float* __restrict__ outF, bf16* __restrict__ outB, int ldo,
                       const float* __restrict__ rowscale,
                       int M, int Kd, int act)
{
    __shared__ bf16 lwt[NT * 16 * 136];          // up to 128 rows x (128+8)
    const int ldp = ldw + 8;
    const int Nd  = NT * 16;

    // cooperative stage WT -> LDS (v8bf chunks; all threads participate)
    {
        const int cpr   = ldw >> 3;              // 8-elem chunks per row
        const int total = Nd * cpr;
        for (int c = threadIdx.x; c < total; c += blockDim.x) {
            const int n  = c / cpr;
            const int k8 = (c - n * cpr) << 3;
            *(v8bf*)(lwt + n * ldp + k8) = *(const v8bf*)(WT + (size_t)n * ldw + k8);
        }
    }
    __syncthreads();

    const int wave = blockIdx.x * (blockDim.x >> 5) + (threadIdx.x >> 5);
    const int lane = threadIdx.x & 31;
    const int m0   = wave * 16;
    if (m0 >= M) return;                          // safe: after the only barrier
    const int half = lane >> 4;
    const int l15  = lane & 15;

    v8f zero8 = {0.f,0.f,0.f,0.f,0.f,0.f,0.f,0.f};
    v8f acc[NT];
#pragma unroll
    for (int i = 0; i < NT; ++i) acc[i] = zero8;

    const bf16* arow = A + (size_t)(m0 + l15) * lda;

    for (int k0 = 0; k0 < Kd; k0 += 32) {
        // A fragment (16-bit A 16x32 layout, ISA 7.12.2)
        Frag16 af;
        const bf16* ap = arow + k0 + half * 8;
        af.h[0] = *(const v8bf*)(ap);
        af.h[1] = *(const v8bf*)(ap + 16);
        if (k0 + 32 < Kd) __builtin_prefetch(arow + k0 + 32, 0, 2); // global_prefetch

        // batch all B fragments from LDS first, then the WMMAs, so the
        // compiler can overlap ds waits with matrix issue
        Frag16 bfr[NT];
#pragma unroll
        for (int nt = 0; nt < NT; ++nt) {
            const bf16* bp = lwt + (nt * 16 + l15) * ldp + k0 + half * 8;
            bfr[nt].h[0] = *(const v8bf*)(bp);
            bfr[nt].h[1] = *(const v8bf*)(bp + 16);
        }
#pragma unroll
        for (int nt = 0; nt < NT; ++nt)
            acc[nt] = __builtin_amdgcn_wmma_f32_16x16x32_bf16(
                false, af.v, false, bfr[nt].v, (short)0, acc[nt], false, false);
    }

    // Epilogue: C/D 32-bit 16x16 layout: lane holds column n = l15,
    // rows r + 8*half for VGPR r = 0..7.
#pragma unroll
    for (int nt = 0; nt < NT; ++nt) {
        const int n  = nt * 16 + l15;
        const float bs = bias ? bias[n] : 0.f;
#pragma unroll
        for (int r = 0; r < 8; ++r) {
            const int row = m0 + r + half * 8;
            float y = acc[nt][r] + bs;
            if (act) y = y / (1.f + __expf(-y));        // SiLU
            if (rowscale) y *= rowscale[row];           // fused cutoff scale
            const size_t o = (size_t)row * ldo + n;
            if (outF) outF[o] = y;
            if (outB) outB[o] = (bf16)y;
        }
    }
}

// Convert + transpose weight: W[Kd][Nd] f32 -> WT[Nd][ldw] bf16 (zero-padded K)
__global__ void k_prep_wt(const float* __restrict__ W, bf16* __restrict__ WT,
                          int Kd, int Nd, int ldw)
{
    int t = blockIdx.x * blockDim.x + threadIdx.x;
    int total = Nd * ldw;
    if (t >= total) return;
    int n = t / ldw, kk = t % ldw;
    WT[t] = (kk < Kd) ? (bf16)W[(size_t)kk * Nd + n] : (bf16)0.f;
}

__global__ void k_zero(float* p, int n)
{
    int i = blockIdx.x * blockDim.x + threadIdx.x;
    if (i < n) p[i] = 0.f;
}

// a = emb[Z] (f32 + bf16 mirrors)
__global__ void k_embed(const int* __restrict__ Z, const float* __restrict__ emb,
                        float* __restrict__ a, bf16* __restrict__ a_bf, int total)
{
    int t = blockIdx.x * blockDim.x + threadIdx.x;
    if (t >= total) return;
    int f = t & (NFc - 1), r = t >> 7;
    float v = emb[(size_t)Z[r] * NFc + f];
    a[t] = v; a_bf[t] = (bf16)v;
}

// Per-edge geometry: V, D, 1/D, polynomial cutoff, Bessel RBF (padded to 32)
__global__ void k_geom(const float* __restrict__ R, const int* __restrict__ N,
                       const float* __restrict__ NM,
                       float* __restrict__ D, float* __restrict__ Dinv,
                       float* __restrict__ co, float* __restrict__ V3,
                       bf16* __restrict__ rbf, int Redge)
{
    int e = blockIdx.x * blockDim.x + threadIdx.x;
    if (e >= Redge) return;
    int i = e >> 6;            // /K
    int b = i >> 8;            // /A
    int j = (b << 8) + N[e];
    float vx = R[(size_t)j*3+0] - R[(size_t)i*3+0];
    float vy = R[(size_t)j*3+1] - R[(size_t)i*3+1];
    float vz = R[(size_t)j*3+2] - R[(size_t)i*3+2];
    float d2 = vx*vx + vy*vy + vz*vz;
    bool valid = (NM[e] > 0.f) && (d2 > 1e-12f);
    float d  = valid ? sqrtf(d2) : 0.f;
    float di = valid ? 1.f / d : 0.f;
    D[e] = d; Dinv[e] = di;
    V3[(size_t)e*3+0] = vx; V3[(size_t)e*3+1] = vy; V3[(size_t)e*3+2] = vz;
    // cutoff p=9: 1 - 55 x^9 + 99 x^10 - 45 x^11, x = d/5
    float x = d * 0.2f;
    float x2 = x*x, x4 = x2*x2, x8 = x4*x4, x9 = x8*x;
    float f = 1.f - 55.f*x9 + 99.f*x9*x - 45.f*x9*x2;
    co[e] = (d < 5.f) ? f : 0.f;
    const float s  = 0.632455532f;   // sqrt(2/5)
    const float w0 = 0.62831853f;    // pi/5
    bf16* rp = rbf + (size_t)e * 32;
#pragma unroll
    for (int n1 = 1; n1 <= RESc; ++n1)
        rp[n1-1] = valid ? (bf16)(s * __sinf(w0 * (float)n1 * d) * di) : (bf16)0.f;
#pragma unroll
    for (int c = RESc; c < 32; ++c) rp[c] = (bf16)0.f;
}

// msij = a_m[i] * a_m[j] * rbf_m  (bf16 out, GEMM input)
__global__ void k_msij(const float* __restrict__ a_m, const int* __restrict__ N,
                       const bf16* __restrict__ rbf_m, bf16* __restrict__ msij,
                       int total)
{
    int t = blockIdx.x * blockDim.x + threadIdx.x;
    if (t >= total) return;
    int f = t & (NFc - 1);
    int e = t >> 7;
    int i = e >> 6, b = i >> 8, j = (b << 8) + N[e];
    float v = a_m[(size_t)i*NFc+f] * a_m[(size_t)j*NFc+f] * (float)rbf_m[t];
    msij[t] = (bf16)v;
}

// NF->1 dense: wave-per-row dot product + shfl_xor wave32 reduction
__global__ void k_dense_vec(const bf16* __restrict__ X, int ldx,
                            const float* __restrict__ w, const float* __restrict__ bias,
                            float* __restrict__ out, float* __restrict__ accum,
                            int M, int Kd)
{
    int wave = blockIdx.x * (blockDim.x >> 5) + (threadIdx.x >> 5);
    int lane = threadIdx.x & 31;
    if (wave >= M) return;
    float s = 0.f;
    for (int c = lane; c < Kd; c += 32)
        s += (float)X[(size_t)wave * ldx + c] * w[c];
    for (int m = 16; m > 0; m >>= 1) s += __shfl_xor(s, m, 32);
    if (lane == 0) {
        float y = s + bias[0];
        out[wave] = y;
        if (accum) accum[wave] += y;
    }
}

// q_dyn += q * qm_out
__global__ void k_qdyn(const float* __restrict__ q, const float* __restrict__ qm,
                       float* __restrict__ q_dyn, int total)
{
    int t = blockIdx.x * blockDim.x + threadIdx.x;
    if (t >= total) return;
    q_dyn[t] += q[t >> 7] * qm[t];
}

// ---------------------------------------------------------------------------
// Fused per-atom edge update + K-reduction + a-update (deterministic).
// One 256-thread block per atom i. Wave w owns edges k = w, w+8, ..., w+56.
// Per edge: b_dyn update, diff = q_i*q_j - b_dyn, force coefficient
// (wave shfl reduction over features), and register accumulation of
// sum_k D_inv*diff per feature. Cross-wave feature combine via 4 KB LDS.
// Replaces the 64 MB dterm write + 64 MB read per iteration.
// ---------------------------------------------------------------------------
__global__ void k_edge_reduce(const float* __restrict__ q_dyn,
                              const float* __restrict__ bij,
                              const bf16*  __restrict__ bm,
                              const float* __restrict__ e_out,
                              const float* __restrict__ Dinv,
                              const int*   __restrict__ N,
                              float* __restrict__ b_dyn,
                              float* __restrict__ fcoef,
                              float* __restrict__ a,
                              bf16*  __restrict__ a_bf)
{
    __shared__ float part[8][NFc];
    const int i    = blockIdx.x;                 // atom row
    const int wave = threadIdx.x >> 5;
    const int lane = threadIdx.x & 31;
    const int b    = i >> 8;

    float qi[4], ei[4], sf[4];
#pragma unroll
    for (int c = 0; c < 4; ++c) {
        const int f = lane + 32 * c;
        qi[c] = q_dyn[(size_t)i * NFc + f];
        ei[c] = e_out[(size_t)i * NFc + f];
        sf[c] = 0.f;
    }

    for (int k = wave; k < Kc; k += 8) {
        const int e  = i * Kc + k;
        const int j  = (b << 8) + N[e];
        const float be = bij[e];
        const float di = Dinv[e];
        float g = 0.f;
#pragma unroll
        for (int c = 0; c < 4; ++c) {
            const int f = lane + 32 * c;
            const size_t idx = (size_t)e * NFc + f;
            const float bd = b_dyn[idx] + be * (float)bm[idx];
            b_dyn[idx] = bd;
            const float diff = qi[c] * q_dyn[(size_t)j * NFc + f] - bd;
            sf[c] += di * diff;
            g     += ei[c] * diff;               // dE/d(D_inv) explicit part
        }
        for (int m = 16; m > 0; m >>= 1) g += __shfl_xor(g, m, 32);
        if (lane == 0) fcoef[e] += g;            // one wave owns edge e
    }

#pragma unroll
    for (int c = 0; c < 4; ++c) part[wave][lane + 32 * c] = sf[c];
    __syncthreads();

    const int t = threadIdx.x;
    if (t < NFc) {
        float s = 0.f;
#pragma unroll
        for (int w = 0; w < 8; ++w) s += part[w][t];
        const size_t idx = (size_t)i * NFc + t;
        const float na = a[idx] + e_out[idx] * s;
        a[idx] = na;
        a_bf[idx] = (bf16)na;
    }
}

// Approximate forces from the explicit D^-1 sensitivity (scatter via f32 atomics)
__global__ void k_force(const float* __restrict__ fcoef, const float* __restrict__ Dinv,
                        const float* __restrict__ V3, const int* __restrict__ N,
                        float* __restrict__ F, int Redge)
{
    int e = blockIdx.x * blockDim.x + threadIdx.x;
    if (e >= Redge) return;
    int i = e >> 6, b = i >> 8, j = (b << 8) + N[e];
    float di = Dinv[e];
    float c = fcoef[e] * di * di * di;   // dE/dD * dD/dR direction scale
    float fx = c * V3[(size_t)e*3+0];
    float fy = c * V3[(size_t)e*3+1];
    float fz = c * V3[(size_t)e*3+2];
    atomicAdd(&F[(size_t)i*3+0], -fx);
    atomicAdd(&F[(size_t)i*3+1], -fy);
    atomicAdd(&F[(size_t)i*3+2], -fz);
    atomicAdd(&F[(size_t)j*3+0],  fx);
    atomicAdd(&F[(size_t)j*3+1],  fy);
    atomicAdd(&F[(size_t)j*3+2],  fz);
}

__global__ void k_write_q(const float* __restrict__ q_lat, const float* __restrict__ AM,
                          float* __restrict__ outQ, int n)
{
    int i = blockIdx.x * blockDim.x + threadIdx.x;
    if (i < n) outQ[i] = q_lat[i] * AM[i];
}

__global__ void k_write_b(const float* __restrict__ b_lat, const float* __restrict__ NM,
                          float* __restrict__ outB, int n)
{
    int e = blockIdx.x * blockDim.x + threadIdx.x;
    if (e < n) outB[e] = b_lat[e] * ((NM[e] != 0.f) ? 1.f : 0.f);
}

// E[b] = sum_a Ei[b,a]*AM[b,a]  (one 256-thread block per batch, A == 256)
__global__ void k_batch_sum(const float* __restrict__ Ei, const float* __restrict__ AM,
                            float* __restrict__ outE)
{
    __shared__ float red[256];
    int b = blockIdx.x, t = threadIdx.x;
    red[t] = Ei[(size_t)b * Ac + t] * AM[(size_t)b * Ac + t];
    __syncthreads();
    for (int s = 128; s > 0; s >>= 1) {
        if (t < s) red[t] += red[t + s];
        __syncthreads();
    }
    if (t == 0) outE[b] = red[0];
}

// ---------------------------------------------------------------------------
// Host orchestration
// ---------------------------------------------------------------------------
extern "C" void kernel_launch(void* const* d_in, const int* in_sizes, int n_in,
                              void* d_out, int out_size, void* d_ws, size_t ws_size,
                              hipStream_t stream)
{
    (void)in_sizes; (void)n_in; (void)out_size; (void)ws_size;

    // Input flattening (setup_inputs insertion order):
    // 0:R 1:Z 2:AM 3:N 4:NM 5:emb, then per-iter 26 arrays
    // (rbf W,b | a0 W,b | a1 W,b | q0 W,b | q1 W,b | qm0 W,b | qm1 W,b |
    //  b0 W,b | b1 W,b | bm0 W,b | bm1 W,b | e0 W,b | e1 W,b), then atom 6.
    auto P = [&](int i) { return (const float*)d_in[i]; };
    const float* R   = P(0);
    const int*   Z   = (const int*)d_in[1];
    const float* AM  = P(2);
    const int*   N   = (const int*)d_in[3];
    const float* NM  = P(4);
    const float* emb = P(5);

    float* out  = (float*)d_out;
    float* outE = out;
    float* outF = out + Bc;
    float* outQ = outF + (size_t)RTOT * 3;
    float* outB = outQ + RTOT;

    // workspace allocator (256B aligned)
    char* ws = (char*)d_ws;
    size_t off = 0;
    auto alloc = [&](size_t bytes) -> char* {
        char* p = ws + off;
        off += (bytes + 255) & ~(size_t)255;
        return p;
    };

    // f32 buffers
    float* a      = (float*)alloc((size_t)RTOT*NFc*4);
    float* a_m    = (float*)alloc((size_t)RTOT*NFc*4);
    float* qm_out = (float*)alloc((size_t)RTOT*NFc*4);
    float* e_out  = (float*)alloc((size_t)RTOT*NFc*4);
    float* q_dyn  = (float*)alloc((size_t)RTOT*NFc*4);
    float* qbuf   = (float*)alloc((size_t)RTOT*4);
    float* q_lat  = (float*)alloc((size_t)RTOT*4);
    float* Ei     = (float*)alloc((size_t)RTOT*4);
    float* Dbuf   = (float*)alloc((size_t)REDGE*4);
    float* Dinv   = (float*)alloc((size_t)REDGE*4);
    float* cobuf  = (float*)alloc((size_t)REDGE*4);
    float* V3     = (float*)alloc((size_t)REDGE*3*4);
    float* bij    = (float*)alloc((size_t)REDGE*4);
    float* b_lat  = (float*)alloc((size_t)REDGE*4);
    float* fcoef  = (float*)alloc((size_t)REDGE*4);
    float* b_dyn  = (float*)alloc((size_t)REDGE*NFc*4);
    // bf16 buffers
    bf16* a_bf   = (bf16*)alloc((size_t)RTOT*NFc*2);
    bf16* ahid   = (bf16*)alloc((size_t)RTOT*NFc*2);
    bf16* h2     = (bf16*)alloc((size_t)RTOT*64*2);
    bf16* rbf_bf = (bf16*)alloc((size_t)REDGE*32*2);
    bf16* rbf_m  = (bf16*)alloc((size_t)REDGE*NFc*2);
    bf16* msij   = (bf16*)alloc((size_t)REDGE*NFc*2);
    bf16* ehid   = (bf16*)alloc((size_t)REDGE*NFc*2);
    bf16* bm_bf  = (bf16*)alloc((size_t)REDGE*NFc*2);

    // weight prep (convert + transpose to bf16)
    auto prep = [&](const float* W, int Kd, int Nd, int ldw) -> bf16* {
        bf16* wt = (bf16*)alloc((size_t)Nd * ldw * 2);
        int total = Nd * ldw;
        k_prep_wt<<<(total + 255) / 256, 256, 0, stream>>>(W, wt, Kd, Nd, ldw);
        return wt;
    };

    struct IterW {
        bf16 *rbf, *a0, *a1, *q0, *qm0, *qm1, *b0, *bm0, *bm1, *e0, *e1;
        const float *b_rbf, *b_a0, *b_a1, *b_q0, *W_q1, *b_q1, *b_qm0, *b_qm1,
                    *b_b0, *W_b1, *b_b1, *b_bm0, *b_bm1, *b_e0, *b_e1;
    } iw[3];

    for (int it = 0; it < 3; ++it) {
        int base = 6 + it * 26;
        iw[it].rbf  = prep(P(base + 0), RESc, NFc, 32); iw[it].b_rbf = P(base + 1);
        iw[it].a0   = prep(P(base + 2), NFc, NFc, NFc); iw[it].b_a0  = P(base + 3);
        iw[it].a1   = prep(P(base + 4), NFc, NFc, NFc); iw[it].b_a1  = P(base + 5);
        iw[it].q0   = prep(P(base + 6), NFc, NFc, NFc); iw[it].b_q0  = P(base + 7);
        iw[it].W_q1 = P(base + 8);                      iw[it].b_q1  = P(base + 9);
        iw[it].qm0  = prep(P(base +10), NFc, NFc, NFc); iw[it].b_qm0 = P(base +11);
        iw[it].qm1  = prep(P(base +12), NFc, NFc, NFc); iw[it].b_qm1 = P(base +13);
        iw[it].b0   = prep(P(base +14), NFc, NFc, NFc); iw[it].b_b0  = P(base +15);
        iw[it].W_b1 = P(base +16);                      iw[it].b_b1  = P(base +17);
        iw[it].bm0  = prep(P(base +18), NFc, NFc, NFc); iw[it].b_bm0 = P(base +19);
        iw[it].bm1  = prep(P(base +20), NFc, NFc, NFc); iw[it].b_bm1 = P(base +21);
        iw[it].e0   = prep(P(base +22), NFc, NFc, NFc); iw[it].b_e0  = P(base +23);
        iw[it].e1   = prep(P(base +24), NFc, NFc, NFc); iw[it].b_e1  = P(base +25);
    }
    bf16* at0 = prep(P(84), NFc, NFc, NFc); const float* b_at0 = P(85);
    bf16* at1 = prep(P(86), NFc, 64, NFc);  const float* b_at1 = P(87);
    const float* W_at2 = P(88);             const float* b_at2 = P(89);

    auto zero = [&](float* p, int n) {
        k_zero<<<(n + 255) / 256, 256, 0, stream>>>(p, n);
    };
    auto gemm8 = [&](const bf16* Aa, int lda, const bf16* WT, int ldw, const float* bias,
                     float* oF, bf16* oB, int ldo, const float* rs, int M, int Kd, int act) {
        int blocks = (M / 16 + 7) / 8;
        k_gemm<8><<<blocks, 256, 0, stream>>>(Aa, lda, WT, ldw, bias, oF, oB, ldo, rs, M, Kd, act);
    };
    auto gemm4 = [&](const bf16* Aa, int lda, const bf16* WT, int ldw, const float* bias,
                     float* oF, bf16* oB, int ldo, const float* rs, int M, int Kd, int act) {
        int blocks = (M / 16 + 7) / 8;
        k_gemm<4><<<blocks, 256, 0, stream>>>(Aa, lda, WT, ldw, bias, oF, oB, ldo, rs, M, Kd, act);
    };
    auto dvec = [&](const bf16* X, int ldx, const float* w, const float* b,
                    float* o, float* acc, int M, int Kd) {
        k_dense_vec<<<(M + 7) / 8, 256, 0, stream>>>(X, ldx, w, b, o, acc, M, Kd);
    };

    // ---- setup ----
    k_geom<<<(REDGE + 255) / 256, 256, 0, stream>>>(R, N, NM, Dbuf, Dinv, cobuf, V3,
                                                    rbf_bf, REDGE);
    k_embed<<<(RTOT * NFc + 255) / 256, 256, 0, stream>>>(Z, emb, a, a_bf, RTOT * NFc);
    zero(q_dyn, RTOT * NFc); zero(q_lat, RTOT);
    zero(b_dyn, REDGE * NFc); zero(b_lat, REDGE); zero(fcoef, REDGE);
    zero(outF, RTOT * 3);

    // ---- interaction iterations ----
    for (int it = 0; it < 3; ++it) {
        const IterW& w = iw[it];
        // rbf_m = dense(rbf) * co   (cutoff fused as rowscale)
        gemm8(rbf_bf, 32, w.rbf, 32, w.b_rbf, nullptr, rbf_m, NFc, cobuf, REDGE, 32, 0);
        // a_m = mlp2(a, p.a)
        gemm8(a_bf, NFc, w.a0, NFc, w.b_a0, nullptr, ahid, NFc, nullptr, RTOT, NFc, 1);
        gemm8(ahid, NFc, w.a1, NFc, w.b_a1, a_m, nullptr, NFc, nullptr, RTOT, NFc, 0);
        // msij = a_m[i] * a_m[j] * rbf_m
        k_msij<<<(REDGE * NFc + 255) / 256, 256, 0, stream>>>(a_m, N, rbf_m, msij,
                                                              REDGE * NFc);
        // q = mlp2(a, p.q); q_lat += q
        gemm8(a_bf, NFc, w.q0, NFc, w.b_q0, nullptr, ahid, NFc, nullptr, RTOT, NFc, 1);
        dvec(ahid, NFc, w.W_q1, w.b_q1, qbuf, q_lat, RTOT, NFc);
        // q_dyn += q * mlp2(a, p.qm)
        gemm8(a_bf, NFc, w.qm0, NFc, w.b_qm0, nullptr, ahid, NFc, nullptr, RTOT, NFc, 1);
        gemm8(ahid, NFc, w.qm1, NFc, w.b_qm1, qm_out, nullptr, NFc, nullptr, RTOT, NFc, 0);
        k_qdyn<<<(RTOT * NFc + 255) / 256, 256, 0, stream>>>(qbuf, qm_out, q_dyn,
                                                             RTOT * NFc);
        // bij = mlp2(msij, p.b); b_lat += bij
        gemm8(msij, NFc, w.b0, NFc, w.b_b0, nullptr, ehid, NFc, nullptr, REDGE, NFc, 1);
        dvec(ehid, NFc, w.W_b1, w.b_b1, bij, b_lat, REDGE, NFc);
        // bm = mlp2(msij, p.bm) -> bf16 (read exactly once downstream)
        gemm8(msij, NFc, w.bm0, NFc, w.b_bm0, nullptr, ehid, NFc, nullptr, REDGE, NFc, 1);
        gemm8(ehid, NFc, w.bm1, NFc, w.b_bm1, nullptr, bm_bf, NFc, nullptr, REDGE, NFc, 0);
        // e_out = mlp2(a, p.e)  (old a)
        gemm8(a_bf, NFc, w.e0, NFc, w.b_e0, nullptr, ahid, NFc, nullptr, RTOT, NFc, 1);
        gemm8(ahid, NFc, w.e1, NFc, w.b_e1, e_out, nullptr, NFc, nullptr, RTOT, NFc, 0);
        // fused edge update + K-reduction + a update (block per atom)
        k_edge_reduce<<<RTOT, 256, 0, stream>>>(q_dyn, bij, bm_bf, e_out, Dinv, N,
                                                b_dyn, fcoef, a, a_bf);
    }

    // ---- outputs ----
    k_write_q<<<(RTOT + 255) / 256, 256, 0, stream>>>(q_lat, AM, outQ, RTOT);
    k_write_b<<<(REDGE + 255) / 256, 256, 0, stream>>>(b_lat, NM, outB, REDGE);
    k_force<<<(REDGE + 255) / 256, 256, 0, stream>>>(fcoef, Dinv, V3, N, outF, REDGE);

    // atom energy head: 128 -> 128(silu) -> 64(silu) -> 1
    gemm8(a_bf, NFc, at0, NFc, b_at0, nullptr, ahid, NFc, nullptr, RTOT, NFc, 1);
    gemm4(ahid, NFc, at1, NFc, b_at1, nullptr, h2, 64, nullptr, RTOT, NFc, 1);
    dvec(h2, 64, W_at2, b_at2, Ei, nullptr, RTOT, 64);
    k_batch_sum<<<Bc, 256, 0, stream>>>(Ei, AM, outE);
}
